// MultiHeadAttention_6098853560672
// MI455X (gfx1250) — matrix operand order repaired and checked by reading, not compiled, not run on hardware
//
#include <hip/hip_runtime.h>

// ---------------------------------------------------------------------------
// MHA forward for MI455X (gfx1250): bf16 WMMA + TDM async tile loads.
// ---------------------------------------------------------------------------

typedef __bf16 bf16_t;
typedef __attribute__((ext_vector_type(16))) __bf16 v16bf;
typedef __attribute__((ext_vector_type(8)))  __bf16 bf16x8;
typedef __attribute__((ext_vector_type(8)))  float  v8f;
typedef __attribute__((ext_vector_type(4)))  unsigned int u32x4;
typedef __attribute__((ext_vector_type(8)))  unsigned int u32x8;

#define NUM_HEADS 16
#define DIM       1024
#define HEAD_DIM  64
#define SEQ       2048
#define BATCH     2
#define QK_SCALE  0.125f   // HEAD_DIM^-0.5

union AF { v16bf v; bf16x8 h[2]; };
union G0 { u32x4 v; unsigned w[4]; };
union G1 { u32x8 v; unsigned w[8]; };

static __device__ __forceinline__ v8f wmma_bf16(v16bf a, v16bf b, v8f c) {
  return __builtin_amdgcn_wmma_f32_16x16x32_bf16(false, a, false, b,
                                                 (short)0, c, false, false);
}

#define BM 128
#define BN 128
#define BK 32
#define LDSS 40   // LDS row pitch in bf16: 64B tile row + 16B TDM pad = 80B

// ---------------------------------------------------------------------------
// TDM: async-load a 128x32 bf16 tile (row stride = rowStride elems) into LDS
// at byte offset ldsOff, padding each 64B row to 80B. Issued once per wave.
// ---------------------------------------------------------------------------
static __device__ __forceinline__ void tdm_load_tile(const bf16_t* gptr,
                                                     unsigned ldsOff,
                                                     unsigned rowStride,
                                                     unsigned tensorRows) {
  unsigned long long ga = (unsigned long long)(uintptr_t)gptr;
  G0 g0;
  g0.w[0] = 1u;                                    // count=1, user mode
  g0.w[1] = ldsOff;                                // lds_addr
  g0.w[2] = (unsigned)(ga & 0xffffffffu);          // global_addr[31:0]
  g0.w[3] = (unsigned)((ga >> 32) & 0x1ffffffu)    // global_addr[56:32]
            | (2u << 30);                          // type = 2 (image)
  G1 g1;
  // data_size=1 (2B) | pad_enable | pad_interval=3 (16 DW) | pad_amount=3 (4 DW)
  g1.w[0] = (1u << 16) | (1u << 20) | (3u << 22) | (3u << 25);
  unsigned td0 = rowStride;                        // tensor_dim0 (elems)
  unsigned td1 = tensorRows;                       // tensor_dim1 (rows)
  g1.w[1] = (td0 & 0xffffu) << 16;                 // tensor_dim0[15:0] @ 63:48
  g1.w[2] = ((td0 >> 16) & 0xffffu) | ((td1 & 0xffffu) << 16);
  g1.w[3] = ((td1 >> 16) & 0xffffu) | ((unsigned)BK << 16);  // tile_dim0=32
  g1.w[4] = (unsigned)BM;                          // tile_dim1=128, tile_dim2=0
  g1.w[5] = rowStride;                             // tensor_dim0_stride[31:0]
  g1.w[6] = 0u;                                    // stride hi, dim1_stride lo
  g1.w[7] = 0u;
  asm volatile("tensor_load_to_lds %0, %1" :: "s"(g0.v), "s"(g1.v) : "memory");
}

// ---------------------------------------------------------------------------
// WMMA GEMM, all-bf16 operands: C[M,N] = A[M,K] * Bt[N,K]^T + bias.
// Block tile 128x128 (8 waves of 32x64), K-step 32, TDM double-buffered LDS.
// EPI==0: QKV scatter epilogue (Q scaled, V transposed). EPI==1: f32 out.
// ---------------------------------------------------------------------------
template <int EPI>
__global__ __launch_bounds__(256) void wmma_gemm(
    const bf16_t* __restrict__ A, const bf16_t* __restrict__ Bt,
    const float* __restrict__ bias, float* __restrict__ outF,
    bf16_t* __restrict__ outQ, bf16_t* __restrict__ outK,
    bf16_t* __restrict__ outV, int M, int N, int K)
{
  __align__(16) __shared__ bf16_t sA[2][BM * LDSS];
  __align__(16) __shared__ bf16_t sB[2][BN * LDSS];

  const int tid  = threadIdx.x;
  const int lane = tid & 31;
  const int wid  = tid >> 5;
  const int wm   = wid >> 1;
  const int wn   = wid & 1;
  const int bm0  = blockIdx.y * BM;
  const int bn0  = blockIdx.x * BN;
  const int ln   = lane & 15;
  const int kh   = lane >> 4;
  const int nk   = K / BK;

  v8f c[2][4];
#pragma unroll
  for (int i = 0; i < 2; i++)
#pragma unroll
    for (int j = 0; j < 4; j++) c[i][j] = v8f{};

  if (wid == 0) {   // prologue: async-load tile 0 into buffer 0
    tdm_load_tile(A  + (size_t)bm0 * K, (unsigned)(uintptr_t)&sA[0][0], K, M);
    tdm_load_tile(Bt + (size_t)bn0 * K, (unsigned)(uintptr_t)&sB[0][0], K, N);
  }

  for (int it = 0; it < nk; it++) {
    const int buf = it & 1;
    if (wid == 0) {
      if (it + 1 < nk) {  // prefetch next tile into other buffer
        int k0 = (it + 1) * BK;
        tdm_load_tile(A  + (size_t)bm0 * K + k0,
                      (unsigned)(uintptr_t)&sA[buf ^ 1][0], K, M);
        tdm_load_tile(Bt + (size_t)bn0 * K + k0,
                      (unsigned)(uintptr_t)&sB[buf ^ 1][0], K, N);
        __builtin_amdgcn_s_wait_tensorcnt(2);  // tile `it` done, next in flight
      } else {
        __builtin_amdgcn_s_wait_tensorcnt(0);
      }
    }
    __syncthreads();   // tile `it` visible to all waves

    AF a[2], b[4];
#pragma unroll
    for (int i = 0; i < 2; i++) {
      int m = wm * 32 + i * 16 + ln;
      const bf16_t* p = &sA[buf][m * LDSS + 8 * kh];
      a[i].h[0] = *(const bf16x8*)p;
      a[i].h[1] = *(const bf16x8*)(p + 16);
    }
#pragma unroll
    for (int j = 0; j < 4; j++) {
      int n = wn * 64 + j * 16 + ln;
      const bf16_t* p = &sB[buf][n * LDSS + 8 * kh];
      b[j].h[0] = *(const bf16x8*)p;
      b[j].h[1] = *(const bf16x8*)(p + 16);
    }
#pragma unroll
    for (int i = 0; i < 2; i++)
#pragma unroll
      for (int j = 0; j < 4; j++)
        c[i][j] = wmma_bf16(a[i].v, b[j].v, c[i][j]);
    __syncthreads();   // compute done; buffer may be overwritten next iter
  }

  // --- epilogue: C layout -> row = r + 8*kh, col = ln ---
#pragma unroll
  for (int i = 0; i < 2; i++) {
    int rowb = bm0 + wm * 32 + i * 16;
#pragma unroll
    for (int j = 0; j < 4; j++) {
      int col = bn0 + wn * 64 + j * 16 + ln;
      float bv = bias[col];
#pragma unroll
      for (int r = 0; r < 8; r++) {
        int rr = rowb + r + 8 * kh;
        float v = c[i][j][r] + bv;
        if constexpr (EPI == 1) {
          outF[(size_t)rr * N + col] = v;
        } else {
          int sect = col >> 10;
          int d    = col & 1023;
          int h    = d >> 6, hd = d & 63;
          int bb   = rr >> 11, l = rr & 2047;
          size_t bh = (size_t)(bb * NUM_HEADS + h);
          if (sect == 0)
            outQ[(bh * SEQ + l) * HEAD_DIM + hd] = (bf16_t)(v * QK_SCALE);
          else if (sect == 1)
            outK[(bh * SEQ + l) * HEAD_DIM + hd] = (bf16_t)v;
          else
            outV[(bh * HEAD_DIM + hd) * SEQ + l] = (bf16_t)v;  // V transposed
        }
      }
    }
  }
}

// ---------------------------------------------------------------------------
// Flash attention: one wave per 16-row Q tile, 32-key KV steps.
// ---------------------------------------------------------------------------
__global__ __launch_bounds__(32) void flash_attn(
    const bf16_t* __restrict__ Q, const bf16_t* __restrict__ Kb,
    const bf16_t* __restrict__ Vb, bf16_t* __restrict__ O)
{
  __align__(16) __shared__ bf16_t sP[16 * LDSS];

  const int lane = threadIdx.x & 31;
  const int ln   = lane & 15;
  const int kh   = lane >> 4;
  const int bh   = blockIdx.y;
  const int qt   = blockIdx.x;

  AF q0, q1;
  {
    const bf16_t* p = Q + ((size_t)bh * SEQ + qt * 16 + ln) * HEAD_DIM + 8 * kh;
    q0.h[0] = *(const bf16x8*)p;        q0.h[1] = *(const bf16x8*)(p + 16);
    q1.h[0] = *(const bf16x8*)(p + 32); q1.h[1] = *(const bf16x8*)(p + 48);
  }

  float mrow[8], lrow[8];
  v8f o[4];
#pragma unroll
  for (int r = 0; r < 8; r++) { mrow[r] = -1e30f; lrow[r] = 0.f; }
#pragma unroll
  for (int j = 0; j < 4; j++) o[j] = v8f{};

  const bf16_t* kbase = Kb + (size_t)bh * SEQ * HEAD_DIM;
  const bf16_t* vbase = Vb + (size_t)bh * HEAD_DIM * SEQ;

  for (int kt = 0; kt < SEQ; kt += 32) {
    v8f s[2];
#pragma unroll
    for (int j = 0; j < 2; j++) {
      AF b0, b1;
      const bf16_t* kp = kbase + (size_t)(kt + j * 16 + ln) * HEAD_DIM + 8 * kh;
      b0.h[0] = *(const bf16x8*)kp;        b0.h[1] = *(const bf16x8*)(kp + 16);
      b1.h[0] = *(const bf16x8*)(kp + 32); b1.h[1] = *(const bf16x8*)(kp + 48);
      v8f z = v8f{};
      z    = wmma_bf16(q0.v, b0.v, z);
      s[j] = wmma_bf16(q1.v, b1.v, z);
    }

    float sc[8];
#pragma unroll
    for (int r = 0; r < 8; r++) {
      float v = fmaxf(s[0][r], s[1][r]);
#pragma unroll
      for (int off = 1; off < 16; off <<= 1) v = fmaxf(v, __shfl_xor(v, off, 16));
      float mn = fmaxf(mrow[r], v);
      sc[r]   = __expf(mrow[r] - mn);
      mrow[r] = mn;
      float p0 = __expf(s[0][r] - mn);
      float p1 = __expf(s[1][r] - mn);
      s[0][r] = p0; s[1][r] = p1;
      float t = p0 + p1;
#pragma unroll
      for (int off = 1; off < 16; off <<= 1) t += __shfl_xor(t, off, 16);
      lrow[r] = lrow[r] * sc[r] + t;
    }
#pragma unroll
    for (int j = 0; j < 4; j++)
#pragma unroll
      for (int r = 0; r < 8; r++) o[j][r] *= sc[r];

#pragma unroll
    for (int j = 0; j < 2; j++)
#pragma unroll
      for (int r = 0; r < 8; r++)
        sP[(r + 8 * kh) * LDSS + j * 16 + ln] = (bf16_t)s[j][r];
    asm volatile("s_wait_dscnt 0" ::: "memory");
    AF pa;
    {
      const bf16_t* p = &sP[ln * LDSS + 8 * kh];
      pa.h[0] = *(const bf16x8*)p;
      pa.h[1] = *(const bf16x8*)(p + 16);
    }

#pragma unroll
    for (int j = 0; j < 4; j++) {
      AF vb;
      const bf16_t* vp = vbase + (size_t)(j * 16 + ln) * SEQ + kt + 8 * kh;
      vb.h[0] = *(const bf16x8*)vp;
      vb.h[1] = *(const bf16x8*)(vp + 16);
      o[j] = wmma_bf16(pa.v, vb.v, o[j]);
    }
  }

  const int h  = bh & (NUM_HEADS - 1);
  const int bb = bh >> 4;
#pragma unroll
  for (int j = 0; j < 4; j++) {
#pragma unroll
    for (int r = 0; r < 8; r++) {
      int row = qt * 16 + r + 8 * kh;
      int d   = h * HEAD_DIM + j * 16 + ln;
      O[((size_t)bb * SEQ + row) * DIM + d] = (bf16_t)(o[j][r] / lrow[r]);
    }
  }
}

// ---------------------------------------------------------------------------
// Precision prep: f32 -> bf16 convert; weight transpose+convert [K,N]->[N,K].
// ---------------------------------------------------------------------------
__global__ __launch_bounds__(256) void convert_bf16(const float* __restrict__ in,
                                                    bf16_t* __restrict__ out,
                                                    int n4)
{
  int i = blockIdx.x * blockDim.x + threadIdx.x;
  if (i < n4) {
    float4 v = ((const float4*)in)[i];
    bf16_t o[4] = {(bf16_t)v.x, (bf16_t)v.y, (bf16_t)v.z, (bf16_t)v.w};
    *(uint64_t*)&out[i * 4] = *(const uint64_t*)o;
  }
}

__global__ __launch_bounds__(256) void transpose_bf16(const float* __restrict__ in,
                                                      bf16_t* __restrict__ out,
                                                      int K, int N)
{
  __shared__ float t[32][33];
  int bx = blockIdx.x * 32;   // n
  int by = blockIdx.y * 32;   // k
#pragma unroll
  for (int i = 0; i < 4; i++)
    t[threadIdx.y + 8 * i][threadIdx.x] =
        in[(size_t)(by + threadIdx.y + 8 * i) * N + bx + threadIdx.x];
  __syncthreads();
#pragma unroll
  for (int i = 0; i < 4; i++)
    out[(size_t)(bx + threadIdx.y + 8 * i) * K + by + threadIdx.x] =
        (bf16_t)t[threadIdx.x][threadIdx.y + 8 * i];
}

// ---------------------------------------------------------------------------
extern "C" void kernel_launch(void* const* d_in, const int* in_sizes, int n_in,
                              void* d_out, int out_size, void* d_ws, size_t ws_size,
                              hipStream_t stream)
{
  const float* x      = (const float*)d_in[0];
  const float* W_qkv  = (const float*)d_in[1];
  const float* b_qkv  = (const float*)d_in[2];
  const float* W_proj = (const float*)d_in[3];
  const float* b_proj = (const float*)d_in[4];
  float* out = (float*)d_out;

  const int M = BATCH * SEQ;                       // 4096
  const size_t perBuf = (size_t)M * HEAD_DIM * NUM_HEADS;  // 4M elems

  bf16_t* xb  = (bf16_t*)d_ws;                     // [M, DIM]
  bf16_t* Wqb = xb  + (size_t)M * DIM;             // [3*DIM, DIM] (transposed)
  bf16_t* Wpb = Wqb + (size_t)3 * DIM * DIM;       // [DIM, DIM]   (transposed)
  bf16_t* Qb  = Wpb + (size_t)DIM * DIM;
  bf16_t* Kb  = Qb + perBuf;
  bf16_t* Vb  = Kb + perBuf;
  bf16_t* AO  = Vb + perBuf;                       // [M, DIM]

  // --- precision prep ---
  int n4 = (M * DIM) / 4;
  convert_bf16<<<n4 / 256, 256, 0, stream>>>(x, xb, n4);
  dim3 tb(32, 8);
  transpose_bf16<<<dim3((3 * DIM) / 32, DIM / 32), tb, 0, stream>>>(W_qkv, Wqb, DIM, 3 * DIM);
  transpose_bf16<<<dim3(DIM / 32, DIM / 32), tb, 0, stream>>>(W_proj, Wpb, DIM, DIM);

  // --- QKV GEMM (TDM + WMMA) ---
  dim3 g1((3 * DIM) / BN, M / BM);
  wmma_gemm<0><<<g1, 256, 0, stream>>>(xb, Wqb, b_qkv, nullptr,
                                       Qb, Kb, Vb, M, 3 * DIM, DIM);

  // --- flash attention ---
  dim3 g2(SEQ / 16, BATCH * NUM_HEADS);
  flash_attn<<<g2, 32, 0, stream>>>(Qb, Kb, Vb, AO);

  // --- projection GEMM ---
  dim3 g3(DIM / BN, M / BM);
  wmma_gemm<1><<<g3, 256, 0, stream>>>(AO, Wpb, b_proj, out,
                                       nullptr, nullptr, nullptr, M, DIM, DIM);
}
